// HierarchicalGraphPooling_6030134083773
// MI455X (gfx1250) — matrix operand order repaired and checked by reading, not compiled, not run on hardware
//
#include <hip/hip_runtime.h>

typedef __attribute__((ext_vector_type(16))) _Float16 v16h;
typedef __attribute__((ext_vector_type(8)))  float    v8f;

#define BB 4
#define NN 2048
#define CC 128
#define EE 4
#define KK 1024   // top-k = N/2
#define H1DIM 64
#define H2DIM 16

// ---------------------------------------------------------------------------
// Kernel 1: score MLP. 8 waves / block; each wave computes a 16-row tile.
// Layer 1 ([16,128] x [128,64]) runs on v_wmma_f32_16x16x32_f16.
// Layers 2/3 are tiny (64->16->1) and run scalar on lanes 0..15 via LDS.
// ---------------------------------------------------------------------------
__global__ __launch_bounds__(256) void mlp_scores_kernel(
    const float* __restrict__ x,  const float* __restrict__ w1, const float* __restrict__ b1,
    const float* __restrict__ w2, const float* __restrict__ b2,
    const float* __restrict__ w3, const float* __restrict__ b3,
    float* __restrict__ scores)
{
    __shared__ float h1s[8][16][H1DIM];     // 32 KB
    const int tid  = threadIdx.x;
    const int wave = tid >> 5;
    const int lane = tid & 31;
    const int half = lane >> 4;             // lane-half selects K sub-block
    const int sub  = lane & 15;             // A: row M ; B/C: column N
    const int tile = blockIdx.x * 8 + wave; // 16-row tile over flat [B*N] rows
    const int row0 = tile * 16;

    // ---- layer 1 via WMMA: A fragments (16x32 f16, ISA A-layout) ----
    v16h afrag[4];
#pragma unroll
    for (int ks = 0; ks < 4; ++ks) {
#pragma unroll
        for (int t = 0; t < 16; ++t) {
            // VGPR j holds K = (j<4 ? 0 : 16) + half*8 + (j%4)*2 + e  (t = 2j+e)
            int kl = (t < 8) ? (half * 8 + t) : (16 + half * 8 + (t - 8));
            afrag[ks][t] = (_Float16)x[(size_t)(row0 + sub) * CC + ks * 32 + kl];
        }
    }
#pragma unroll
    for (int nt = 0; nt < 4; ++nt) {        // four 16-wide column tiles of H1
        v8f acc = {};
#pragma unroll
        for (int ks = 0; ks < 4; ++ks) {    // K = 128 in steps of 32
            v16h bf;                        // B 32x16 f16: lane=col N, t -> K
#pragma unroll
            for (int t = 0; t < 16; ++t) {
                int kr = ks * 32 + ((t < 8) ? (half * 8 + t) : (16 + half * 8 + (t - 8)));
                bf[t] = (_Float16)w1[(size_t)kr * H1DIM + nt * 16 + sub];
            }
            acc = __builtin_amdgcn_wmma_f32_16x16x32_f16(
                false, afrag[ks], false, bf, (short)0, acc, false, false);
        }
        // C layout: lane holds col N=sub, VGPR r -> row M = half*8 + r
        float bias = b1[nt * 16 + sub];
#pragma unroll
        for (int r = 0; r < 8; ++r) {
            float v = acc[r] + bias;
            h1s[wave][half * 8 + r][nt * 16 + sub] = v > 0.f ? v : 0.f;
        }
    }
    __syncthreads();

    // ---- layers 2/3 scalar (1040 MACs per row) ----
    if (lane < 16) {
        const int row = lane;
        float s = b3[0];
#pragma unroll 4
        for (int o = 0; o < H2DIM; ++o) {
            float a2 = b2[o];
#pragma unroll 8
            for (int c = 0; c < H1DIM; ++c)
                a2 += h1s[wave][row][c] * w2[c * H2DIM + o];
            s += (a2 > 0.f ? a2 : 0.f) * w3[o];
        }
        scores[row0 + row] = s;
    }
}

// ---------------------------------------------------------------------------
// Kernel 2: stable-descending top-k by rank counting (matches jax.lax.top_k
// ordering incl. ties). Scores staged in LDS; broadcast inner loop.
// ---------------------------------------------------------------------------
__global__ __launch_bounds__(256) void topk_rank_kernel(
    const float* __restrict__ scores, int* __restrict__ idx)
{
    __shared__ float ss[NN];
    const int b   = blockIdx.y;
    const int tid = threadIdx.x;
#pragma unroll
    for (int t = 0; t < NN / 256; ++t)
        ss[tid + 256 * t] = scores[b * NN + tid + 256 * t];
    __syncthreads();

    const int   i  = blockIdx.x * 256 + tid;
    const float my = ss[i];
    int rank = 0;
    for (int j = 0; j < NN; ++j) {
        float sj = ss[j];
        rank += (sj > my) || (sj == my && j < i);
    }
    if (rank < KK) idx[b * KK + rank] = i;
}

// ---------------------------------------------------------------------------
// Kernel 3: pooled_features[b,i,:] = x[b, idx[b,i], :]   (float4 path)
// ---------------------------------------------------------------------------
__global__ __launch_bounds__(256) void gather_feat_kernel(
    const float4* __restrict__ x4, const int* __restrict__ idx,
    float4* __restrict__ out4)
{
    const int g  = blockIdx.x * 256 + threadIdx.x;   // B*KK*(CC/4)
    const int c4 = g & 31;                           // CC/4 = 32
    const int i  = (g >> 5) & (KK - 1);
    const int b  = g >> 15;                          // 5 + 10
    const int r  = idx[b * KK + i];
    out4[g] = x4[((size_t)b * NN + r) * (CC / 4) + c4];
}

// ---------------------------------------------------------------------------
// Kernel 4: pooled_adjacency[b,i,j] = adj[b, idx[i], idx[j]]
// ---------------------------------------------------------------------------
__global__ __launch_bounds__(256) void gather_adj_kernel(
    const float* __restrict__ adj, const int* __restrict__ idx,
    float* __restrict__ out)
{
    const size_t g = (size_t)blockIdx.x * 256 + threadIdx.x;  // B*KK*KK
    const int j = (int)(g & (KK - 1));
    const int i = (int)((g >> 10) & (KK - 1));
    const int b = (int)(g >> 20);
    const int ri = idx[b * KK + i];
    const int rj = idx[b * KK + j];
    out[g] = adj[((size_t)b * NN + ri) * NN + rj];
}

// ---------------------------------------------------------------------------
// Kernel 5: pooled_edge_features[b,i,j,:] = ef[b, idx[i], idx[j], :]
// E=4 => one float4 per (i,j); coalesced writes, L2-resident scattered reads.
// ---------------------------------------------------------------------------
__global__ __launch_bounds__(256) void gather_edge_kernel(
    const float4* __restrict__ ef4, const int* __restrict__ idx,
    float4* __restrict__ out4)
{
    const size_t g = (size_t)blockIdx.x * 256 + threadIdx.x;  // B*KK*KK
    const int j = (int)(g & (KK - 1));
    const int i = (int)((g >> 10) & (KK - 1));
    const int b = (int)(g >> 20);
    const int ri = idx[b * KK + i];
    const int rj = idx[b * KK + j];
    out4[g] = ef4[((size_t)b * NN + ri) * NN + rj];
}

// ---------------------------------------------------------------------------
extern "C" void kernel_launch(void* const* d_in, const int* in_sizes, int n_in,
                              void* d_out, int out_size, void* d_ws, size_t ws_size,
                              hipStream_t stream)
{
    const float* x   = (const float*)d_in[0];
    const float* adj = (const float*)d_in[1];
    const float* ef  = (const float*)d_in[2];
    // d_in[3] = superpoint_centroids (unused by the reference)
    const float* w1  = (const float*)d_in[4];
    const float* b1  = (const float*)d_in[5];
    const float* w2  = (const float*)d_in[6];
    const float* b2  = (const float*)d_in[7];
    const float* w3  = (const float*)d_in[8];
    const float* b3  = (const float*)d_in[9];

    float* scores = (float*)d_ws;                                   // B*N f32 (32 KB)
    int*   idx    = (int*)((char*)d_ws + (size_t)BB * NN * sizeof(float)); // B*k i32 (16 KB)

    float* pf = (float*)d_out;                       // [B,k,C]
    float* pa = pf + (size_t)BB * KK * CC;           // [B,k,k]
    float* pe = pa + (size_t)BB * KK * KK;           // [B,k,k,E]

    mlp_scores_kernel<<<BB * NN / 128, 256, 0, stream>>>(x, w1, b1, w2, b2, w3, b3, scores);
    topk_rank_kernel<<<dim3(NN / 256, BB), 256, 0, stream>>>(scores, idx);
    gather_feat_kernel<<<(BB * KK * (CC / 4)) / 256, 256, 0, stream>>>(
        (const float4*)x, idx, (float4*)pf);
    gather_adj_kernel<<<((size_t)BB * KK * KK) / 256, 256, 0, stream>>>(adj, idx, pa);
    gather_edge_kernel<<<((size_t)BB * KK * KK) / 256, 256, 0, stream>>>(
        (const float4*)ef, idx, (float4*)pe);
}
